// ModulatedConv3d_63239098466931
// MI455X (gfx1250) — matrix operand order repaired and checked by reading, not compiled, not run on hardware
//
#include <hip/hip_runtime.h>
#include <stdint.h>

#define BB   8
#define CIN  512
#define COUT 512
#define KW   3
#define SDIM 512
#define PP   2048
#define PG   (PP + 2)   // guarded length: zero rows at pg==0 and pg==PP+1

typedef __attribute__((ext_vector_type(16))) __bf16        v16bf;
typedef __attribute__((ext_vector_type(8)))  float         v8f;
typedef __attribute__((ext_vector_type(8)))  unsigned short ushort8;

union FragBF {
  ushort8 h[2];
  v16bf   v;
};

__device__ __forceinline__ unsigned short f32_to_bf16(float f) {
  unsigned int u = __float_as_uint(f);
  u += 0x7FFFu + ((u >> 16) & 1u);      // round to nearest even
  return (unsigned short)(u >> 16);
}

// -------- kernel 1: s[b,ci] = style[b,:] . mod_weight[ci,:] * (1/sqrt(512)) + mod_bias[ci]
__global__ void k_style_mod(const float* __restrict__ style,
                            const float* __restrict__ mod_weight,
                            const float* __restrict__ mod_bias,
                            float* __restrict__ s) {
  const int ci = blockIdx.x * blockDim.x + threadIdx.x;   // 0..511
  const int b  = blockIdx.y;
  const float mod_scale = 0.04419417382415922f;           // 1/sqrt(512)
  const float* st = style + (size_t)b * SDIM;
  const float* mw = mod_weight + (size_t)ci * SDIM;
  float acc = 0.f;
  for (int d = 0; d < SDIM; ++d) acc += st[d] * mw[d];
  s[(size_t)b * CIN + ci] = acc * mod_scale + mod_bias[ci];
}

// -------- kernel 2: demod per (b,co), fold into bf16 weights wmod[b][k][co][ci]
__global__ void k_demod_weights(const float* __restrict__ weight,   // [COUT][CIN][KW]
                                const float* __restrict__ s,        // [B][CIN]
                                unsigned short* __restrict__ wmod)  // [B][KW][COUT][CIN]
{
  const int b   = blockIdx.x / COUT;
  const int co  = blockIdx.x % COUT;
  const int tid = threadIdx.x;                            // 256 threads
  const float conv_scale = 0.02551551815399144f;          // 1/sqrt(512*3)

  __shared__ float red[256];
  float partial = 0.f;
  for (int ci = tid; ci < CIN; ci += 256) {
    const float sv = s[(size_t)b * CIN + ci];
    const float* wp = weight + ((size_t)co * CIN + ci) * KW;
#pragma unroll
    for (int k = 0; k < KW; ++k) {
      float w = conv_scale * wp[k] * sv;
      partial += w * w;
    }
  }
  red[tid] = partial;
  __syncthreads();
  for (int off = 128; off > 0; off >>= 1) {
    if (tid < off) red[tid] += red[tid + off];
    __syncthreads();
  }
  const float demod = rsqrtf(red[0] + 1e-8f);

  for (int ci = tid; ci < CIN; ci += 256) {
    const float sv = s[(size_t)b * CIN + ci];
    const float* wp = weight + ((size_t)co * CIN + ci) * KW;
#pragma unroll
    for (int k = 0; k < KW; ++k) {
      float w = conv_scale * wp[k] * sv * demod;
      wmod[(((size_t)b * KW + k) * COUT + co) * CIN + ci] = f32_to_bf16(w);
    }
  }
}

// -------- kernel 3: xT[b][pg][ci] = bf16(input[b][ci][pg-1]), zero guard rows
__global__ void k_xpose_bf16(const float* __restrict__ input,      // [B][CIN][PP]
                             unsigned short* __restrict__ xt)      // [B][PG][CIN]
{
  const size_t t = (size_t)blockIdx.x * blockDim.x + threadIdx.x;
  const int ci = (int)(t % CIN);
  const size_t r = t / CIN;
  const int pg = (int)(r % PG);
  const int b  = (int)(r / PG);
  float v = 0.f;
  if (pg >= 1 && pg <= PP)
    v = input[((size_t)b * CIN + ci) * PP + (pg - 1)];
  xt[((size_t)b * PG + pg) * CIN + ci] = f32_to_bf16(v);
}

// -------- kernel 4: WMMA GEMM conv, software-pipelined (double-buffered fragments).
// One wave -> 16(M=co) x 64(N=p) output tile; 48 steps (3 taps x 16 K-chunks).
#define NSUB  4
#define NSTEP (KW * (CIN / 32))   // 48

struct Tile {
  FragBF A;
  FragBF Bv[NSUB];
};

__device__ __forceinline__ void load_step(const unsigned short* __restrict__ wmod_b,
                                          const unsigned short* __restrict__ xt_b,
                                          int co0, int p0, int l, int half,
                                          int step, Tile& T) {
  const int k = step >> 4;      // tap
  const int c = step & 15;      // K-chunk of 32
  // A fragment: lane l holds M = co0+l; halves pick K subranges 0..7 / 8..15 (+16)
  const unsigned short* aBase =
      wmod_b + ((size_t)k * COUT + co0 + l) * CIN + half * 8 + c * 32;
  T.A.h[0] = *(const ushort8*)(aBase);
  T.A.h[1] = *(const ushort8*)(aBase + 16);
  // B fragment: halves pick K subranges 0..7 / 8..15 (+16); row = guarded p index (p+k-1 -> +k)
  const unsigned short* bBase =
      xt_b + ((size_t)(p0 + l + k)) * CIN + half * 8 + c * 32;
#pragma unroll
  for (int t = 0; t < NSUB; ++t) {
    const unsigned short* bb = bBase + (size_t)t * 16 * CIN;
    T.Bv[t].h[0] = *(const ushort8*)(bb);
    T.Bv[t].h[1] = *(const ushort8*)(bb + 16);
  }
}

__global__ void __launch_bounds__(256)
k_wmma_conv(const unsigned short* __restrict__ wmod,   // [B][KW][COUT][CIN] bf16
            const unsigned short* __restrict__ xt,     // [B][PG][CIN]       bf16
            float* __restrict__ out)                   // [B][COUT][PP]      f32
{
  const int wave = blockIdx.x * (blockDim.x >> 5) + (threadIdx.x >> 5);
  const int lane = threadIdx.x & 31;
  const int l    = lane & 15;
  const int half = lane >> 4;

  const int pT  = wave & 31;          // 32 supertiles of 64 along P
  const int coT = (wave >> 5) & 31;   // 32 tiles of 16 along Cout
  const int b   = wave >> 10;         // 8 batches

  const int p0  = pT * 64;
  const int co0 = coT * 16;

  const unsigned short* wmod_b = wmod + (size_t)b * KW * COUT * CIN;
  const unsigned short* xt_b   = xt + (size_t)b * PG * CIN;

  v8f acc[NSUB] = {};

  Tile Tbuf[2];
  load_step(wmod_b, xt_b, co0, p0, l, half, 0, Tbuf[0]);

#pragma unroll
  for (int step = 0; step < NSTEP; ++step) {
    const int cur = step & 1;
    // prefetch next step's fragments into the alternate buffer (overlaps WMMAs)
    if (step + 1 < NSTEP)
      load_step(wmod_b, xt_b, co0, p0, l, half, step + 1, Tbuf[cur ^ 1]);
#pragma unroll
    for (int t = 0; t < NSUB; ++t) {
      acc[t] = __builtin_amdgcn_wmma_f32_16x16x32_bf16(
          /*neg_a=*/false, Tbuf[cur].A.v, /*neg_b=*/false, Tbuf[cur].Bv[t].v,
          /*c_mod=*/(short)0, acc[t], /*reuse_a=*/false, /*reuse_b=*/false);
    }
  }

  // C/D layout: VGPR r -> M = half*8 + r, N = l
  float* op = out + ((size_t)b * COUT + co0 + half * 8) * PP + p0 + l;
#pragma unroll
  for (int t = 0; t < NSUB; ++t) {
#pragma unroll
    for (int r = 0; r < 8; ++r)
      op[(size_t)r * PP + t * 16] = acc[t][r];
  }
}

extern "C" void kernel_launch(void* const* d_in, const int* in_sizes, int n_in,
                              void* d_out, int out_size, void* d_ws, size_t ws_size,
                              hipStream_t stream) {
  const float* input      = (const float*)d_in[0];  // [B][CIN][PP]
  const float* style      = (const float*)d_in[1];  // [B][SDIM]
  const float* weight     = (const float*)d_in[2];  // [1][COUT][CIN][KW]
  const float* mod_weight = (const float*)d_in[3];  // [CIN][SDIM]
  const float* mod_bias   = (const float*)d_in[4];  // [CIN]
  float* out = (float*)d_out;

  char* ws = (char*)d_ws;
  float*          s    = (float*)(ws + 0);                 // 16 KB
  unsigned short* wmod = (unsigned short*)(ws + 32768);    // 12.6 MB
  unsigned short* xt   = (unsigned short*)(ws + 12615680); // 16.8 MB

  k_style_mod    <<<dim3(CIN / 256, BB), 256, 0, stream>>>(style, mod_weight, mod_bias, s);
  k_demod_weights<<<BB * COUT, 256, 0, stream>>>(weight, s, wmod);
  k_xpose_bf16   <<<(BB * PG * CIN) / 256, 256, 0, stream>>>(input, xt);
  k_wmma_conv    <<<1024, 256, 0, stream>>>(wmod, xt, out);   // 8192 waves, 1 tile each
}
